// DecoderBeam_49898930045513
// MI455X (gfx1250) — matrix operand order repaired and checked by reading, not compiled, not run on hardware
//
#include <hip/hip_runtime.h>
#include <hip/hip_bf16.h>
#include <math.h>

typedef __attribute__((ext_vector_type(16))) __bf16 v16bf;
typedef __attribute__((ext_vector_type(8)))  float  v8f;

#define B_   32
#define S_   1024
#define U_   1024
#define EMB_ 256
#define VOC_ 32000
#define EXT_ 32500
#define GIN_ (U_ + EMB_)   /* 1280 */
#define G3_  (3 * U_)      /* 3072 */

#define A_STRIDE  1032     /* halves per LDS A row: 1024 + 8 pad (16B-aligned reads) */
#define BT_STRIDE 72       /* halves per LDS Bt row: 64 + 8 pad */

__device__ __forceinline__ unsigned pack_bf2(float lo, float hi) {
  union { __bf16 h[2]; unsigned u; } v;
  v.h[0] = (__bf16)lo; v.h[1] = (__bf16)hi;   // native v_cvt bf16 path
  return v.u;
}
__device__ __forceinline__ float sigmoidf_(float x) { return 1.0f / (1.0f + __expf(-x)); }

// ---------------------------------------------------------------------------
// Kernel 0: hb[b,u] = hidden[b,:] @ aW2[:,u] + ab1[u] + ab2[u]   (32x1024)
// ---------------------------------------------------------------------------
__global__ void hb_kernel(const float* __restrict__ hidden, const float* __restrict__ aW2,
                          const float* __restrict__ ab1, const float* __restrict__ ab2,
                          float* __restrict__ hb) {
  const int idx = blockIdx.x * blockDim.x + threadIdx.x;    // 32768
  const int b = idx >> 10, u = idx & 1023;
  const float* hrow = hidden + (size_t)b * U_;
  float acc = 0.f;
  for (int k = 0; k < U_; ++k) acc = fmaf(hrow[k], aW2[(size_t)k * U_ + u], acc);
  hb[idx] = acc + ab1[u] + ab2[u];
}

// ---------------------------------------------------------------------------
// Kernel A: fused attention score via WMMA bf16.
// score[b,s] = sum_u tanh( (enc@aW1)[b,s,u] + hb[b,u] ) * aV[u] + abV
// Block = 256 threads = 8 waves; owns b = blk/32, rows s0..s0+31.
// Macro-tile per barrier pair: 32 rows x 128 cols x 64 K -> 4 WMMAs / wave.
// Wave (w): rowgrp = w&1 (16 rows), colgrp = w>>1 (32 of 128 cols -> 2 tiles).
// ---------------------------------------------------------------------------
__global__ void attn_score_kernel(const float* __restrict__ enc, const float* __restrict__ aW1,
                                  const float* __restrict__ hb, const float* __restrict__ aV,
                                  const float* __restrict__ abV, float* __restrict__ score) {
  extern __shared__ char smem_raw[];
  float* hb_s    = (float*)smem_raw;                   // 1024
  float* av_s    = hb_s + U_;                          // 1024
  float* score_s = av_s + U_;                          // 32
  unsigned short* ldsA  = (unsigned short*)(score_s + 32);   // 32 * A_STRIDE
  unsigned short* ldsBt = ldsA + 32 * A_STRIDE;              // 128 * BT_STRIDE

  const int t = threadIdx.x;
  const int wave = t >> 5, lane = t & 31;
  const int half = lane >> 4, l = lane & 15;
  const int rowgrp = wave & 1;
  const int colgrp = wave >> 1;

  const int b  = blockIdx.x >> 5;
  const int s0 = (blockIdx.x & 31) * 32;

  for (int i = t; i < U_; i += 256) { hb_s[i] = hb[(size_t)b * U_ + i]; av_s[i] = aV[i]; }
  if (t < 32) score_s[t] = 0.f;

  // Stage full A slab: 32 s-rows x 1024 K, fp32 -> bf16 in LDS (done once).
  {
    const int row = t >> 3;
    const int kb0 = (t & 7) * 128;
    const float* src = enc + ((size_t)(b * S_ + s0 + row)) * U_;
    unsigned short* dst = ldsA + row * A_STRIDE;
    for (int j = 0; j < 32; ++j) {
      const int kb = kb0 + j * 4;
      float4 v = *(const float4*)(src + kb);
      uint2 p;
      p.x = pack_bf2(v.x, v.y);
      p.y = pack_bf2(v.z, v.w);
      *(uint2*)(dst + kb) = p;
    }
  }
  __syncthreads();

  float partial[8];
  #pragma unroll
  for (int i = 0; i < 8; ++i) partial[i] = 0.f;

  const int arow = rowgrp * 16 + l;   // A rows are 0..15 for both lane-halves

  // B staging assignment (per thread): k-quad kq = (t>>4)*4 (64 K), n-seg = (t&15)*8 (128 N)
  const int kq   = (t >> 4) * 4;
  const int nseg = (t & 15) * 8;

  for (int N0 = 0; N0 < U_; N0 += 128) {
    v8f acc0 = {}, acc1 = {};
    for (int K0 = 0; K0 < U_; K0 += 64) {
      __syncthreads();
      { // stage Bt[n][k] (n in [0,128), k in [0,64)) transposed from aW1, fp32 -> bf16
        float4 r[4][2];
        #pragma unroll
        for (int dk = 0; dk < 4; ++dk) {
          const float* src = aW1 + (size_t)(K0 + kq + dk) * U_ + N0 + nseg;
          r[dk][0] = *(const float4*)(src);
          r[dk][1] = *(const float4*)(src + 4);
        }
        #pragma unroll
        for (int j = 0; j < 8; ++j) {
          const float v0 = ((const float*)&r[0][0])[j];
          const float v1 = ((const float*)&r[1][0])[j];
          const float v2 = ((const float*)&r[2][0])[j];
          const float v3 = ((const float*)&r[3][0])[j];
          uint2 p; p.x = pack_bf2(v0, v1); p.y = pack_bf2(v2, v3);
          *(uint2*)(ldsBt + (nseg + j) * BT_STRIDE + kq) = p;   // ds_store_b64
        }
        if (K0 + 64 < U_)   // global_prefetch_b8 on next K block
          __builtin_prefetch(aW1 + (size_t)(K0 + 64 + kq) * U_ + N0 + nseg, 0, 1);
      }
      __syncthreads();
      // Fragments in contiguous k order (k = 16*half + s), same for A and B:
      // A (16x32): lane<16 -> row l, K 0..15 ; lane>=16 -> row l, K 16..31
      // B (32x16): lane<16 -> col l, K 0..15 ;  lane>=16 -> col l, K 16..31
      const int bn0 = colgrp * 32 + l;
      #pragma unroll
      for (int ks = 0; ks < 64; ks += 32) {
        union { v16bf v; uint4 q[2]; } af, bf0, bf1;
        const int ka = K0 + ks + half * 16;
        af.q[0] = *(const uint4*)(ldsA + arow * A_STRIDE + ka);
        af.q[1] = *(const uint4*)(ldsA + arow * A_STRIDE + ka + 8);
        bf0.q[0] = *(const uint4*)(ldsBt + bn0 * BT_STRIDE + ks + half * 16);
        bf0.q[1] = *(const uint4*)(ldsBt + bn0 * BT_STRIDE + ks + half * 16 + 8);
        bf1.q[0] = *(const uint4*)(ldsBt + (bn0 + 16) * BT_STRIDE + ks + half * 16);
        bf1.q[1] = *(const uint4*)(ldsBt + (bn0 + 16) * BT_STRIDE + ks + half * 16 + 8);
        acc0 = __builtin_amdgcn_wmma_f32_16x16x32_bf16(false, af.v, false, bf0.v,
                                                       (short)0, acc0, false, false);
        acc1 = __builtin_amdgcn_wmma_f32_16x16x32_bf16(false, af.v, false, bf1.v,
                                                       (short)0, acc1, false, false);
      }
    }
    // Fuse: tanh(enc@aW1 + hb) * aV, accumulate per-row partials (both tiles).
    const int col0 = N0 + colgrp * 32 + l;
    const int col1 = col0 + 16;
    const float hb0 = hb_s[col0], av0 = av_s[col0];
    const float hb1 = hb_s[col1], av1 = av_s[col1];
    #pragma unroll
    for (int i = 0; i < 8; ++i) {
      partial[i] = fmaf(tanhf(acc0[i] + hb0), av0, partial[i]);
      partial[i] = fmaf(tanhf(acc1[i] + hb1), av1, partial[i]);
    }
  }

  // Reduce over the 16 lanes of each half (the 16 cols of each tile).
  #pragma unroll
  for (int i = 0; i < 8; ++i) {
    float p = partial[i];
    p += __shfl_xor(p, 1);
    p += __shfl_xor(p, 2);
    p += __shfl_xor(p, 4);
    p += __shfl_xor(p, 8);
    partial[i] = p;
  }
  if (l == 0) {   // one lane per half per wave commits its 8 rows
    #pragma unroll
    for (int i = 0; i < 8; ++i)
      atomicAdd(&score_s[rowgrp * 16 + half * 8 + i], partial[i]);
  }
  __syncthreads();
  if (t < 32) score[(size_t)b * S_ + s0 + t] = score_s[t] + abV[0];
}

// ---------------------------------------------------------------------------
// Kernel B: softmax over S + context = att-weighted sum of enc_out. 1 block/b.
// ---------------------------------------------------------------------------
__global__ void softmax_context_kernel(const float* __restrict__ score, const float* __restrict__ enc,
                                       float* __restrict__ att_out, float* __restrict__ context) {
  __shared__ float att_s[S_];
  __shared__ float red[256];
  const int b = blockIdx.x, t = threadIdx.x;
  const float* srow = score + (size_t)b * S_;
  float m = -1e30f;
  for (int s = t; s < S_; s += 256) m = fmaxf(m, srow[s]);
  red[t] = m; __syncthreads();
  for (int o = 128; o > 0; o >>= 1) { if (t < o) red[t] = fmaxf(red[t], red[t + o]); __syncthreads(); }
  m = red[0]; __syncthreads();
  float lsum = 0.f;
  for (int s = t; s < S_; s += 256) { float e = __expf(srow[s] - m); att_s[s] = e; lsum += e; }
  red[t] = lsum; __syncthreads();
  for (int o = 128; o > 0; o >>= 1) { if (t < o) red[t] += red[t + o]; __syncthreads(); }
  const float inv = 1.0f / red[0];
  __syncthreads();
  for (int s = t; s < S_; s += 256) {
    float a = att_s[s] * inv; att_s[s] = a; att_out[(size_t)b * S_ + s] = a;
  }
  __syncthreads();
  float a0 = 0.f, a1 = 0.f, a2 = 0.f, a3 = 0.f;
  const float* eb = enc + (size_t)b * S_ * U_;
  for (int s = 0; s < S_; ++s) {
    const float a = att_s[s];
    const float* er = eb + (size_t)s * U_;
    a0 = fmaf(a, er[t], a0);
    a1 = fmaf(a, er[t + 256], a1);
    a2 = fmaf(a, er[t + 512], a2);
    a3 = fmaf(a, er[t + 768], a3);
  }
  context[(size_t)b * U_ + t]       = a0;
  context[(size_t)b * U_ + t + 256] = a1;
  context[(size_t)b * U_ + t + 512] = a2;
  context[(size_t)b * U_ + t + 768] = a3;
}

// ---------------------------------------------------------------------------
// Kernel C: GRU step (h0 == 0, so gR path collapses to gB[1]) + p_gen. 1 block/b.
// ---------------------------------------------------------------------------
__global__ void gru_pgen_kernel(const int* __restrict__ x, const float* __restrict__ emb,
                                const float* __restrict__ context, const float* __restrict__ gK,
                                const float* __restrict__ gB, const float* __restrict__ whW,
                                const float* __restrict__ whb, const float* __restrict__ wsW,
                                const float* __restrict__ wsb, const float* __restrict__ wxW,
                                const float* __restrict__ wxb, float* __restrict__ state_out,
                                float* __restrict__ pgen_out) {
  __shared__ float gin[GIN_];
  __shared__ float red[256];
  const int b = blockIdx.x, t = threadIdx.x;
  for (int i = t; i < U_; i += 256) gin[i] = context[(size_t)b * U_ + i];
  const int xid = x[b];
  if (t < EMB_) gin[U_ + t] = emb[(size_t)xid * EMB_ + t];
  __syncthreads();
  float plocal = 0.f;
  for (int jj = 0; jj < 4; ++jj) {
    const int u = t + jj * 256;
    float xz = gB[u], xr = gB[U_ + u], xh = gB[2 * U_ + u];
    for (int k = 0; k < GIN_; ++k) {
      const float g = gin[k];
      const float* gr = gK + (size_t)k * G3_;
      xz = fmaf(g, gr[u], xz);
      xr = fmaf(g, gr[U_ + u], xr);
      xh = fmaf(g, gr[2 * U_ + u], xh);
    }
    const float hz = gB[G3_ + u], hr = gB[G3_ + U_ + u], hh = gB[G3_ + 2 * U_ + u];
    const float z  = sigmoidf_(xz + hz);
    const float r  = sigmoidf_(xr + hr);
    const float hc = tanhf(xh + r * hh);
    const float st = (1.0f - z) * hc;             // state = z*0 + (1-z)*hcand
    state_out[(size_t)b * U_ + u] = st;
    plocal = fmaf(gin[u], whW[u], plocal);        // context . whW
    plocal = fmaf(st,     wsW[u], plocal);        // state . wsW
  }
  red[t] = plocal; __syncthreads();
  for (int o = 128; o > 0; o >>= 1) { if (t < o) red[t] += red[t + o]; __syncthreads(); }
  if (t == 0) {
    float p = red[0] + whb[0] + wsb[0] + (float)xid * wxW[0] + wxb[0];
    pgen_out[b] = sigmoidf_(p);
  }
}

// ---------------------------------------------------------------------------
// Kernel D: final[:, :32000] = p_gen * (state @ fcW + fcb); zero the tail.
// state (128KB) lives in dynamic LDS, broadcast-read; memory-bound on fcW.
// ---------------------------------------------------------------------------
__global__ void pred_final_kernel(const float* __restrict__ state, const float* __restrict__ pgen,
                                  const float* __restrict__ fcW, const float* __restrict__ fcb,
                                  float* __restrict__ final_out) {
  extern __shared__ char smem_raw[];
  float* st_s = (float*)smem_raw;          // 32*1024
  float* pg_s = st_s + B_ * U_;            // 32
  const int t = threadIdx.x;
  for (int i = t; i < B_ * U_; i += 256) st_s[i] = state[i];
  if (t < B_) pg_s[t] = pgen[t];
  __syncthreads();
  const int col = blockIdx.x * 256 + t;
  if (col >= EXT_) return;
  if (col >= VOC_) {
    for (int b = 0; b < B_; ++b) final_out[(size_t)b * EXT_ + col] = 0.f;
    return;
  }
  float acc[B_];
  #pragma unroll
  for (int b = 0; b < B_; ++b) acc[b] = 0.f;
  for (int k = 0; k < U_; ++k) {
    const float w = fcW[(size_t)k * VOC_ + col];
    #pragma unroll
    for (int b = 0; b < B_; ++b) acc[b] = fmaf(st_s[b * U_ + k], w, acc[b]);
  }
  const float bias = fcb[col];
  for (int b = 0; b < B_; ++b)
    final_out[(size_t)b * EXT_ + col] = pg_s[b] * (acc[b] + bias);
}

// ---------------------------------------------------------------------------
// Kernel E: final[b, src_ids[s]] += (1 - p_gen[b]) * att[b, s]  (atomic: dup ids)
// ---------------------------------------------------------------------------
__global__ void scatter_kernel(const int* __restrict__ src_ids, const float* __restrict__ att,
                               const float* __restrict__ pgen, float* __restrict__ final_out) {
  const int b = blockIdx.x, t = threadIdx.x;
  const float c = 1.0f - pgen[b];
  for (int s = t; s < S_; s += 256)
    atomicAdd(&final_out[(size_t)b * EXT_ + src_ids[s]], c * att[(size_t)b * S_ + s]);
}

// ---------------------------------------------------------------------------
extern "C" void kernel_launch(void* const* d_in, const int* in_sizes, int n_in,
                              void* d_out, int out_size, void* d_ws, size_t ws_size,
                              hipStream_t stream) {
  (void)in_sizes; (void)n_in; (void)out_size; (void)ws_size;
  const int*   x      = (const int*)  d_in[0];
  const float* hidden = (const float*)d_in[1];
  const float* enc    = (const float*)d_in[2];
  const int*   srcids = (const int*)  d_in[3];
  /* d_in[4] = ext_vocab scalar (== EXT_) */
  const float* emb = (const float*)d_in[5];
  const float* aW1 = (const float*)d_in[6];
  const float* ab1 = (const float*)d_in[7];
  const float* aW2 = (const float*)d_in[8];
  const float* ab2 = (const float*)d_in[9];
  const float* aV  = (const float*)d_in[10];
  const float* abV = (const float*)d_in[11];
  const float* gK  = (const float*)d_in[12];
  /* d_in[13] = gR (unused: h0 == 0) */
  const float* gB  = (const float*)d_in[14];
  const float* fcW = (const float*)d_in[15];
  const float* fcb = (const float*)d_in[16];
  const float* whW = (const float*)d_in[17];
  const float* whb = (const float*)d_in[18];
  const float* wsW = (const float*)d_in[19];
  const float* wsb = (const float*)d_in[20];
  const float* wxW = (const float*)d_in[21];
  const float* wxb = (const float*)d_in[22];

  float* out = (float*)d_out;
  float* out_final = out;                                  // 32*32500
  float* out_state = out + (size_t)B_ * EXT_;              // 32*1024
  float* out_att   = out_state + B_ * U_;                  // 32*1024
  float* out_pgen  = out_att + B_ * S_;                    // 32

  float* ws      = (float*)d_ws;
  float* hb      = ws;               // 32768 floats
  float* score   = ws + 32768;       // 32768 floats
  float* context = ws + 65536;       // 32768 floats

  hb_kernel<<<128, 256, 0, stream>>>(hidden, aW2, ab1, ab2, hb);

  const size_t lds_a = (size_t)(2 * U_ + 32) * sizeof(float)
                     + (size_t)(32 * A_STRIDE + 128 * BT_STRIDE) * sizeof(unsigned short);
  attn_score_kernel<<<B_ * (S_ / 32), 256, lds_a, stream>>>(enc, aW1, hb, aV, abV, score);

  softmax_context_kernel<<<B_, 256, 0, stream>>>(score, enc, out_att, context);

  gru_pgen_kernel<<<B_, 256, 0, stream>>>(x, emb, context, gK, gB, whW, whb, wsW, wsb,
                                          wxW, wxb, out_state, out_pgen);

  const size_t lds_d = (size_t)(B_ * U_ + B_) * sizeof(float);
  pred_final_kernel<<<(EXT_ + 255) / 256, 256, lds_d, stream>>>(out_state, out_pgen, fcW, fcb, out_final);

  scatter_kernel<<<B_, 256, 0, stream>>>(srcids, out_att, out_pgen, out_final);
}
